// FARGAN_50062138802935
// MI455X (gfx1250) — matrix operand order repaired and checked by reading, not compiled, not run on hardware
//
#include <hip/hip_runtime.h>

// ---------------------------------------------------------------------------
// FARGAN vocoder for gfx1250 (MI455X).
// Phase 1: cond_net precompute (parallel over B x frames).
// Phase 2: persistent scan kernel, 256 blocks x 16 batch rows, 200 steps,
//          all GEMMs via v_wmma_f32_16x16x32_f16, state resident in LDS.
// ---------------------------------------------------------------------------

typedef __attribute__((ext_vector_type(16))) _Float16 v16h;
typedef __attribute__((ext_vector_type(8)))  float    v8f;

union Frag { v16h v; unsigned u[8]; };

__device__ __forceinline__ float sigm(float x) { return 1.0f / (1.0f + __expf(-x)); }
__device__ __forceinline__ float clip1f(float x) { return fminf(1.0f, fmaxf(-1.0f, x)); }

// One 16x16 output tile of  acc += A(16 x Kpad, f16, LDS, row-stride lda)
//                               *  W^T tile  (W is [N x Kpad] f16, global/L2)
// A fragment follows the documented 16-bit A-matrix layout (lane=M, K halves
// split by lane>>4); B fragment: lane = N, same K order as A.
__device__ __forceinline__ v8f gemm16(const _Float16* __restrict__ As, int lda,
                                      const _Float16* __restrict__ W, int Kpad,
                                      int nAbs, int lane, v8f acc)
{
    const int m   = lane & 15;
    const int kh8 = (lane >> 4) << 3;   // A: K-half offset
    const int kb  = (lane >> 4) << 4;   // unused; B takes A's offsets (kh8)
    const _Float16* arow = As + m * lda;
    const _Float16* wrow = W + (size_t)(nAbs + (lane & 15)) * Kpad;
    for (int k0 = 0; k0 < Kpad; k0 += 32) {
        Frag a, b;
#pragma unroll
        for (int p = 0; p < 8; ++p) {
            int ka = k0 + kh8 + ((p < 4) ? (p << 1) : (16 + ((p - 4) << 1)));
            a.u[p] = *(const unsigned*)(arow + ka);
            b.u[p] = *(const unsigned*)(wrow + ka);
        }
        acc = __builtin_amdgcn_wmma_f32_16x16x32_f16(false, a.v, false, b.v,
                                                     (short)0, acc, false, false);
    }
    return acc;
}

// ---------------------------------------------------------------------------
// Weight convert + pad (f32 -> f16, zero pad to Npad x Kpad)
// ---------------------------------------------------------------------------
__global__ void cvt_kernel(const float* __restrict__ src, _Float16* __restrict__ dst,
                           int N, int K, int Kpad, int total)
{
    int i = blockIdx.x * 256 + threadIdx.x;
    if (i >= total) return;
    int n = i / Kpad, k = i % Kpad;
    dst[i] = (n < N && k < K) ? (_Float16)src[n * K + k] : (_Float16)0.0f;
}

// ---------------------------------------------------------------------------
// cond_net stage 1: x1[b,t,:] = tanh(concat(f, pembed) @ fdense1.T), t in 0..51
// ---------------------------------------------------------------------------
__global__ void cond1_kernel(const float* __restrict__ feat, const int* __restrict__ period,
                             const float* __restrict__ pembed, const float* __restrict__ fdense1,
                             _Float16* __restrict__ x1)
{
    int bt = blockIdx.x;            // b*52 + t
    int b = bt / 52, t = bt % 52;
    __shared__ float xin[32];
    int tid = threadIdx.x;
    if (tid < 20) xin[tid] = feat[((size_t)b * 54 + (t + 2)) * 20 + tid];
    if (tid >= 20 && tid < 32) {
        int p = period[b * 54 + t + 2] - 32;
        xin[tid] = pembed[p * 12 + (tid - 20)];
    }
    __syncthreads();
    float acc = 0.0f;
    const float* w = fdense1 + tid * 32;
#pragma unroll
    for (int i = 0; i < 32; ++i) acc += w[i] * xin[i];
    x1[(size_t)bt * 64 + tid] = (_Float16)tanhf(acc);
}

// ---------------------------------------------------------------------------
// cond_net stage 2: conv(k=3) + tanh, then fdense2 + tanh -> cond[b,t,320] f16
// ---------------------------------------------------------------------------
__global__ void cond2_kernel(const _Float16* __restrict__ x1, const float* __restrict__ fconv1,
                             const float* __restrict__ fdense2, _Float16* __restrict__ cond)
{
    int bt = blockIdx.x;            // b*50 + t
    int b = bt / 50, t = bt % 50;
    __shared__ float xw[192];
    __shared__ float xc[128];
    int tid = threadIdx.x;          // 128 threads
    for (int i = tid; i < 192; i += 128) {
        int dt = i >> 6, ii = i & 63;
        xw[i] = (float)x1[((size_t)b * 52 + t + dt) * 64 + ii];
    }
    __syncthreads();
    float acc = 0.0f;
    const float* w = fconv1 + (size_t)tid * 64 * 3;   // [O=128][I=64][k=3]
    for (int i = 0; i < 64; ++i)
#pragma unroll
        for (int dt = 0; dt < 3; ++dt) acc += w[i * 3 + dt] * xw[dt * 64 + i];
    xc[tid] = tanhf(acc);
    __syncthreads();
    for (int o = tid; o < 320; o += 128) {
        float a2 = 0.0f;
        const float* w2 = fdense2 + (size_t)o * 128;
        for (int i = 0; i < 128; ++i) a2 += w2[i] * xc[i];
        cond[(size_t)bt * 320 + o] = (_Float16)tanhf(a2);
    }
}

// ---------------------------------------------------------------------------
// Persistent scan kernel
// ---------------------------------------------------------------------------
struct ScanArgs {
    const _Float16* cond;      // [B*50*320] f16
    const int*      period;    // [B*54]
    const float*    cgw;       // cond_gain_w [80]
    const float*    cgb;       // cond_gain_b [1]
    const _Float16* fwc0;      // [192 x 352]
    const _Float16* fwc0glu;   // [192 x 192]
    const _Float16* g1ih;      // [480 x 288]
    const _Float16* g1hh;      // [480 x 160]
    const _Float16* g1glu;     // [160 x 160]
    const _Float16* g2ih;      // [384 x 256]
    const _Float16* g2hh;      // [384 x 128]
    const _Float16* g2glu;     // [128 x 128]
    const _Float16* g3ih;      // [384 x 224]
    const _Float16* g3hh;      // [384 x 128]
    const _Float16* g3glu;     // [128 x 128]
    const _Float16* skipw;     // [128 x 704]
    const _Float16* skipglu;   // [128 x 128]
    const _Float16* sigw;      // [48  x 128]
    const float*    gow;       // gain_out_w [4 x 192]
    const float*    gob;       // gain_out_b [4]
    float*          out;       // [B x 8000]
};

#define XA_W 704
#define SMEM_BYTES 103424

// GLU: gout = f(t * sigmoid(t @ Wg.T)), t f16 in tb (stride tbld)
__device__ __forceinline__ void glu_stage(int H, const _Float16* tb, int tbld,
                                          const _Float16* Wg, _Float16* gout, int gold,
                                          int wave, int lane, bool outerclip)
{
    const int kh8 = (lane >> 4) << 3;
    for (int jt = wave; jt < (H >> 4); jt += 8) {
        v8f acc = {};
        acc = gemm16(tb, tbld, Wg, H, jt * 16, lane, acc);
        int col = jt * 16 + (lane & 15);
#pragma unroll
        for (int r = 0; r < 8; ++r) {
            float t = (float)tb[(kh8 + r) * tbld + col];
            float v = t * sigm(acc[r]);
            if (outerclip) v = clip1f(v);
            gout[(kh8 + r) * gold + col] = (_Float16)v;
        }
    }
    __syncthreads();
}

// GRU cell: 6 accumulators per j-tile, fused nonlinearity, h update in LDS,
// clipped h written to tb (stride 192) for the following GLU.
__device__ __forceinline__ void gru_stage(int H, int Kx, const _Float16* xA, int hoff,
                                          const _Float16* Wih, const _Float16* Whh,
                                          float* h, _Float16* tb, int wave, int lane)
{
    const int kh8 = (lane >> 4) << 3;
    const int ntile = H >> 4;
    float hn0[8], hn1[8];
    const bool has1 = (wave + 8) < ntile;
    {
        int jt = wave;   // ntile >= 8 always (10/8/8)
        v8f xr = {}, xz = {}, xn = {}, hr = {}, hz = {}, hq = {};
        xr = gemm16(xA, XA_W, Wih, Kx,            jt * 16, lane, xr);
        xz = gemm16(xA, XA_W, Wih, Kx,     H    + jt * 16, lane, xz);
        xn = gemm16(xA, XA_W, Wih, Kx, 2 * H    + jt * 16, lane, xn);
        hr = gemm16(xA + hoff, XA_W, Whh, H,        jt * 16, lane, hr);
        hz = gemm16(xA + hoff, XA_W, Whh, H, H    + jt * 16, lane, hz);
        hq = gemm16(xA + hoff, XA_W, Whh, H, 2 * H + jt * 16, lane, hq);
        int col = jt * 16 + (lane & 15);
#pragma unroll
        for (int r = 0; r < 8; ++r) {
            float rr = sigm(xr[r] + hr[r]);
            float zz = sigm(xz[r] + hz[r]);
            float nn = tanhf(xn[r] + rr * hq[r]);
            hn0[r] = (1.0f - zz) * nn + zz * h[(kh8 + r) * H + col];
        }
    }
    if (has1) {
        int jt = wave + 8;
        v8f xr = {}, xz = {}, xn = {}, hr = {}, hz = {}, hq = {};
        xr = gemm16(xA, XA_W, Wih, Kx,            jt * 16, lane, xr);
        xz = gemm16(xA, XA_W, Wih, Kx,     H    + jt * 16, lane, xz);
        xn = gemm16(xA, XA_W, Wih, Kx, 2 * H    + jt * 16, lane, xn);
        hr = gemm16(xA + hoff, XA_W, Whh, H,        jt * 16, lane, hr);
        hz = gemm16(xA + hoff, XA_W, Whh, H, H    + jt * 16, lane, hz);
        hq = gemm16(xA + hoff, XA_W, Whh, H, 2 * H + jt * 16, lane, hq);
        int col = jt * 16 + (lane & 15);
#pragma unroll
        for (int r = 0; r < 8; ++r) {
            float rr = sigm(xr[r] + hr[r]);
            float zz = sigm(xz[r] + hz[r]);
            float nn = tanhf(xn[r] + rr * hq[r]);
            hn1[r] = (1.0f - zz) * nn + zz * h[(kh8 + r) * H + col];
        }
    }
    __syncthreads();   // everyone done READING h
    {
        int col = wave * 16 + (lane & 15);
#pragma unroll
        for (int r = 0; r < 8; ++r) {
            float v = hn0[r];
            h[(kh8 + r) * H + col] = v;
            tb[(kh8 + r) * 192 + col] = (_Float16)clip1f(v);
        }
    }
    if (has1) {
        int col = (wave + 8) * 16 + (lane & 15);
#pragma unroll
        for (int r = 0; r < 8; ++r) {
            float v = hn1[r];
            h[(kh8 + r) * H + col] = v;
            tb[(kh8 + r) * 192 + col] = (_Float16)clip1f(v);
        }
    }
    __syncthreads();
}

__global__ void __launch_bounds__(256, 1) fargan_scan(ScanArgs args)
{
    extern __shared__ char smem_raw[];
    float*    exc     = (float*)smem_raw;          // 16 x 256 (circular)
    float*    h1      = exc + 16 * 256;            // 16 x 160
    float*    h2      = h1 + 16 * 160;             // 16 x 128
    float*    h3      = h2 + 16 * 128;             // 16 x 128
    float*    gainv   = h3 + 16 * 128;             // 16
    float*    pgv     = gainv + 16;                // 16 x 4
    int*      pitchv  = (int*)(pgv + 64);          // 16
    _Float16* xA      = (_Float16*)(pitchv + 16);  // 16 x 704 (A staging)
    _Float16* tb      = xA + 16 * XA_W;            // 16 x 192 (pre-GLU)
    _Float16* tmpprev = tb + 16 * 192;             // 16 x 164 (fwc_st)
    _Float16* fpb     = tmpprev + 16 * 164;        // 16 x 40 (fpitch)
    _Float16* pvb     = fpb + 16 * 40;             // 16 x 40 (prev)
    _Float16* fwcb    = pvb + 16 * 40;             // 16 x 192
    _Float16* g1b     = fwcb + 16 * 192;           // 16 x 160
    _Float16* g2b     = g1b + 16 * 160;            // 16 x 128
    _Float16* g3b     = g2b + 16 * 128;            // 16 x 128
    _Float16* skb     = g3b + 16 * 128;            // 16 x 128

    const int tid  = threadIdx.x;
    const int wave = tid >> 5;
    const int lane = tid & 31;
    const int kh8  = (lane >> 4) << 3;
    const int b0   = blockIdx.x * 16;

    // zero all state
    for (int i = tid; i < SMEM_BYTES / 4; i += 256) ((unsigned*)smem_raw)[i] = 0u;
    __syncthreads();

    int head = 0;   // circular head of exc (logical i -> phys (head+i)&255)

    for (int s = 0; s < 200; ++s) {
        const int frame = s >> 2, part = s & 3;

        // ---- a0: xcat[:,0:164] = fwc_st (previous tmp) ----
        for (int i = tid; i < 16 * 164; i += 256) {
            int r = i / 164, c = i % 164;
            xA[r * XA_W + c] = tmpprev[i];
        }
        __syncthreads();

        // ---- a1: c80, gain, pitch, K-pad zeros ----
        for (int i = tid; i < 16 * 80; i += 256) {
            int r = i / 80, j = i % 80;
            float v = clip1f((float)args.cond[((size_t)(b0 + r) * 50 + frame) * 320 + part * 80 + j]);
            xA[r * XA_W + 164 + j] = (_Float16)v;
            tmpprev[r * 164 + j]   = (_Float16)v;
        }
        if (tid < 16) {
            int r = tid;
            float acc = args.cgb[0];
            const _Float16* cp = args.cond + ((size_t)(b0 + r) * 50 + frame) * 320 + part * 80;
            for (int j = 0; j < 80; ++j) acc += clip1f((float)cp[j]) * args.cgw[j];
            gainv[r]  = __expf(acc);
            pitchv[r] = args.period[(b0 + r) * 54 + 3 + frame];
        }
        for (int i = tid; i < 16 * 24; i += 256) {   // zero xcat pad 328..351
            int r = i / 24, c = 328 + i % 24;
            xA[r * XA_W + c] = (_Float16)0.0f;
        }
        __syncthreads();

        // ---- a2: pred (pitch-lagged excitation) and prev ----
        for (int i = tid; i < 16 * 44; i += 256) {
            int r = i / 44, j = i % 44;
            int p = pitchv[r];
            int idx = 256 - p + j - 2;
            if (idx >= 256) idx -= p;
            float e  = exc[r * 256 + ((head + idx) & 255)];
            float pr = clip1f(e / (1e-5f + gainv[r]));
            xA[r * XA_W + 244 + j]    = (_Float16)pr;
            tmpprev[r * 164 + 80 + j] = (_Float16)pr;
            if (j >= 2 && j < 42) fpb[r * 40 + (j - 2)] = (_Float16)pr;
        }
        for (int i = tid; i < 16 * 40; i += 256) {
            int r = i / 40, j = i % 40;
            float e  = exc[r * 256 + ((head + 216 + j) & 255)];
            float pv = clip1f(e / (1e-5f + gainv[r]));
            xA[r * XA_W + 288 + j]     = (_Float16)pv;
            tmpprev[r * 164 + 124 + j] = (_Float16)pv;
            pvb[r * 40 + j]            = (_Float16)pv;
        }
        __syncthreads();

        // ---- fwc = clip(glu(tanh(xcat @ fwc0_w.T))) ----
        for (int jt = wave; jt < 12; jt += 8) {
            v8f acc = {};
            acc = gemm16(xA, XA_W, args.fwc0, 352, jt * 16, lane, acc);
            int col = jt * 16 + (lane & 15);
#pragma unroll
            for (int r = 0; r < 8; ++r) tb[(kh8 + r) * 192 + col] = (_Float16)tanhf(acc[r]);
        }
        __syncthreads();
        glu_stage(192, tb, 192, args.fwc0glu, fwcb, 192, wave, lane, true);

        // ---- pg = sigmoid(fwc @ gain_out_w.T + b), N=4 serial dots ----
        if (tid < 64) {
            int r = tid >> 2, o = tid & 3;
            float acc = args.gob[o];
            const float* w = args.gow + o * 192;
            const _Float16* f = fwcb + r * 192;
            for (int i = 0; i < 192; ++i) acc += w[i] * (float)f[i];
            pgv[r * 4 + o] = sigm(acc);
        }
        __syncthreads();

        // ---- GRU1: in = [fwc(192) | pg0*fp(40) | prev(40) | pad16], h @ 288 ----
        for (int i = tid; i < 16 * 288; i += 256) {
            int r = i / 288, c = i % 288;
            _Float16 v;
            if (c < 192)      v = fwcb[r * 192 + c];
            else if (c < 232) v = (_Float16)(pgv[r * 4 + 0] * (float)fpb[r * 40 + (c - 192)]);
            else if (c < 272) v = pvb[r * 40 + (c - 232)];
            else              v = (_Float16)0.0f;
            xA[r * XA_W + c] = v;
        }
        for (int i = tid; i < 16 * 160; i += 256) {
            int r = i / 160, c = i % 160;
            xA[r * XA_W + 288 + c] = (_Float16)h1[r * 160 + c];
        }
        __syncthreads();
        gru_stage(160, 288, xA, 288, args.g1ih, args.g1hh, h1, tb, wave, lane);
        glu_stage(160, tb, 192, args.g1glu, g1b, 160, wave, lane, true);

        // ---- GRU2: in = [g1(160) | pg1*fp(40) | prev(40) | pad16], h @ 256 ----
        for (int i = tid; i < 16 * 256; i += 256) {
            int r = i / 256, c = i % 256;
            _Float16 v;
            if (c < 160)      v = g1b[r * 160 + c];
            else if (c < 200) v = (_Float16)(pgv[r * 4 + 1] * (float)fpb[r * 40 + (c - 160)]);
            else if (c < 240) v = pvb[r * 40 + (c - 200)];
            else              v = (_Float16)0.0f;
            xA[r * XA_W + c] = v;
        }
        for (int i = tid; i < 16 * 128; i += 256) {
            int r = i / 128, c = i % 128;
            xA[r * XA_W + 256 + c] = (_Float16)h2[r * 128 + c];
        }
        __syncthreads();
        gru_stage(128, 256, xA, 256, args.g2ih, args.g2hh, h2, tb, wave, lane);
        glu_stage(128, tb, 192, args.g2glu, g2b, 128, wave, lane, true);

        // ---- GRU3: in = [g2(128) | pg2*fp(40) | prev(40) | pad16], h @ 224 ----
        for (int i = tid; i < 16 * 224; i += 256) {
            int r = i / 224, c = i % 224;
            _Float16 v;
            if (c < 128)      v = g2b[r * 128 + c];
            else if (c < 168) v = (_Float16)(pgv[r * 4 + 2] * (float)fpb[r * 40 + (c - 128)]);
            else if (c < 208) v = pvb[r * 40 + (c - 168)];
            else              v = (_Float16)0.0f;
            xA[r * XA_W + c] = v;
        }
        for (int i = tid; i < 16 * 128; i += 256) {
            int r = i / 128, c = i % 128;
            xA[r * XA_W + 224 + c] = (_Float16)h3[r * 128 + c];
        }
        __syncthreads();
        gru_stage(128, 224, xA, 224, args.g3ih, args.g3hh, h3, tb, wave, lane);
        glu_stage(128, tb, 192, args.g3glu, g3b, 128, wave, lane, true);

        // ---- skip: in = [g1|g2|g3|fwc|pg3*fp|prev|pad16] = 704 ----
        for (int i = tid; i < 16 * 704; i += 256) {
            int r = i / 704, c = i % 704;
            _Float16 v;
            if (c < 160)      v = g1b[r * 160 + c];
            else if (c < 288) v = g2b[r * 128 + (c - 160)];
            else if (c < 416) v = g3b[r * 128 + (c - 288)];
            else if (c < 608) v = fwcb[r * 192 + (c - 416)];
            else if (c < 648) v = (_Float16)(pgv[r * 4 + 3] * (float)fpb[r * 40 + (c - 608)]);
            else if (c < 688) v = pvb[r * 40 + (c - 648)];
            else              v = (_Float16)0.0f;
            xA[r * XA_W + c] = v;
        }
        __syncthreads();
        for (int jt = wave; jt < 8; jt += 8) {
            v8f acc = {};
            acc = gemm16(xA, XA_W, args.skipw, 704, jt * 16, lane, acc);
            int col = jt * 16 + (lane & 15);
#pragma unroll
            for (int r = 0; r < 8; ++r)
                tb[(kh8 + r) * 192 + col] = (_Float16)clip1f(tanhf(acc[r]));
        }
        __syncthreads();
        glu_stage(128, tb, 192, args.skipglu, skb, 128, wave, lane, false);  // no outer clip

        // ---- sig = tanh(skip @ sig_w.T) * gain; append to exc ring + output ----
        const int nh = (head + 40) & 255;
        for (int jt = wave; jt < 3; jt += 8) {   // N padded 40 -> 48
            v8f acc = {};
            acc = gemm16(skb, 128, args.sigw, 128, jt * 16, lane, acc);
            int col = jt * 16 + (lane & 15);
            if (col < 40) {
#pragma unroll
                for (int r = 0; r < 8; ++r) {
                    int m2 = kh8 + r;
                    float v = tanhf(acc[r]) * gainv[m2];
                    exc[m2 * 256 + ((nh + 216 + col) & 255)] = v;
                    args.out[(size_t)(b0 + m2) * 8000 + s * 40 + col] = v;
                }
            }
        }
        __syncthreads();
        head = nh;
    }
}

// ---------------------------------------------------------------------------
// Host launch
// ---------------------------------------------------------------------------
extern "C" void kernel_launch(void* const* d_in, const int* in_sizes, int n_in,
                              void* d_out, int out_size, void* d_ws, size_t ws_size,
                              hipStream_t stream)
{
    (void)in_sizes; (void)n_in; (void)out_size; (void)ws_size;
    const int B = 4096;

    const float* feat    = (const float*)d_in[0];
    const int*   period  = (const int*)d_in[1];
    const float* pembed  = (const float*)d_in[3];
    const float* fdense1 = (const float*)d_in[4];
    const float* fconv1  = (const float*)d_in[5];
    const float* fdense2 = (const float*)d_in[6];
    const float* cgw     = (const float*)d_in[7];
    const float* cgb     = (const float*)d_in[8];
    const float* w_fwc0    = (const float*)d_in[9];
    const float* w_fwc0glu = (const float*)d_in[10];
    const float* w_g1ih  = (const float*)d_in[11];
    const float* w_g1hh  = (const float*)d_in[12];
    const float* w_g1glu = (const float*)d_in[13];
    const float* w_g2ih  = (const float*)d_in[14];
    const float* w_g2hh  = (const float*)d_in[15];
    const float* w_g2glu = (const float*)d_in[16];
    const float* w_g3ih  = (const float*)d_in[17];
    const float* w_g3hh  = (const float*)d_in[18];
    const float* w_g3glu = (const float*)d_in[19];
    const float* w_skip    = (const float*)d_in[20];
    const float* w_skipglu = (const float*)d_in[21];
    const float* w_sig   = (const float*)d_in[22];
    const float* gow     = (const float*)d_in[23];
    const float* gob     = (const float*)d_in[24];

    char* wsb = (char*)d_ws;
    size_t off = 0;
    auto alloc = [&](size_t bytes) -> char* {
        char* p = wsb + off;
        off = (off + bytes + 255) & ~(size_t)255;
        return p;
    };

    _Float16* cond = (_Float16*)alloc((size_t)B * 50 * 320 * 2);
    _Float16* x1   = (_Float16*)alloc((size_t)B * 52 * 64 * 2);

    auto cvtw = [&](const float* src, int N, int K, int Npad, int Kpad) -> _Float16* {
        size_t total = (size_t)Npad * Kpad;
        _Float16* dst = (_Float16*)alloc(total * 2);
        cvt_kernel<<<dim3((unsigned)((total + 255) / 256)), dim3(256), 0, stream>>>(
            src, dst, N, K, Kpad, (int)total);
        return dst;
    };

    ScanArgs a;
    a.cond = cond; a.period = period; a.cgw = cgw; a.cgb = cgb;
    a.fwc0    = cvtw(w_fwc0,    192, 328, 192, 352);
    a.fwc0glu = cvtw(w_fwc0glu, 192, 192, 192, 192);
    a.g1ih    = cvtw(w_g1ih,    480, 272, 480, 288);
    a.g1hh    = cvtw(w_g1hh,    480, 160, 480, 160);
    a.g1glu   = cvtw(w_g1glu,   160, 160, 160, 160);
    a.g2ih    = cvtw(w_g2ih,    384, 240, 384, 256);
    a.g2hh    = cvtw(w_g2hh,    384, 128, 384, 128);
    a.g2glu   = cvtw(w_g2glu,   128, 128, 128, 128);
    a.g3ih    = cvtw(w_g3ih,    384, 208, 384, 224);
    a.g3hh    = cvtw(w_g3hh,    384, 128, 384, 128);
    a.g3glu   = cvtw(w_g3glu,   128, 128, 128, 128);
    a.skipw   = cvtw(w_skip,    128, 688, 128, 704);
    a.skipglu = cvtw(w_skipglu, 128, 128, 128, 128);
    a.sigw    = cvtw(w_sig,      40, 128,  48, 128);
    a.gow = gow; a.gob = gob;
    a.out = (float*)d_out;

    // cond_net precompute
    cond1_kernel<<<dim3(B * 52), dim3(64), 0, stream>>>(feat, period, pembed, fdense1, x1);
    cond2_kernel<<<dim3(B * 50), dim3(128), 0, stream>>>(x1, fconv1, fdense2, cond);

    // persistent WMMA scan: 256 blocks x 16 batch rows, 200 sequential steps
    fargan_scan<<<dim3(B / 16), dim3(256), SMEM_BYTES, stream>>>(a);
}
